// GIN_16990890622985
// MI455X (gfx1250) — matrix operand order repaired and checked by reading, not compiled, not run on hardware
//
#include <hip/hip_runtime.h>
#include <hip/hip_bf16.h>

// ---------------------------------------------------------------------------
// GIN on MI455X (gfx1250, wave32):
//  - message passing: gather + L2-resident f32 atomics (bandwidth bound)
//  - MLP GEMMs: bf16 WMMA (v_wmma_f32_16x16x32_bf16), f32 accumulate,
//    double-buffered GLOBAL_LOAD_ASYNC_TO_LDS_B128 tile pipeline
//  - BatchNorm: atomic column reduction + fused normalize/relu
// ---------------------------------------------------------------------------

typedef __attribute__((ext_vector_type(16))) __bf16 v16bf;
typedef __attribute__((ext_vector_type(8)))  float  v8f;
typedef __attribute__((ext_vector_type(4)))  float  fv4;
typedef __attribute__((ext_vector_type(4)))  __bf16 bv4;
typedef __attribute__((ext_vector_type(4)))  int    v4i;
typedef unsigned short bfraw;   // raw bf16 bits

union FragAB { uint4 q[2]; v16bf v; };   // 32B = 8 VGPRs of bf16 operand data

#ifndef __has_builtin
#define __has_builtin(x) 0
#endif
#if __has_builtin(__builtin_amdgcn_global_load_async_to_lds_b128) && \
    __has_builtin(__builtin_amdgcn_s_wait_asynccnt)
#define HAVE_ASYNC_LDS 1
#else
#define HAVE_ASYNC_LDS 0
#endif

#if HAVE_ASYNC_LDS
typedef __attribute__((address_space(1))) v4i v4i_g;   // global (AS1) int4
typedef __attribute__((address_space(3))) v4i v4i_l;   // LDS    (AS3) int4
__device__ __forceinline__ void async_cp16(const void* g, void* l) {
    __builtin_amdgcn_global_load_async_to_lds_b128(
        (v4i_g*)const_cast<void*>(g), (v4i_l*)l, 0, 0);
}
#endif

__device__ __forceinline__ bfraw f2bf(float f) {
    return __builtin_bit_cast(unsigned short, (__bf16)f);   // v_cvt_*bf16_f32
}

// --------------------------- small helper kernels ---------------------------

__global__ void zero_f4(float4* __restrict__ p, long long n4) {
    long long i = (long long)blockIdx.x * blockDim.x + threadIdx.x;
    if (i < n4) p[i] = make_float4(0.f, 0.f, 0.f, 0.f);
}

// packed f32 -> bf16 conversion, 4 elements/thread (v_cvt_pk_bf16_f32)
__global__ void f32_to_bf16_vec(const float* __restrict__ src, bfraw* __restrict__ dst,
                                long long n4) {
    long long i = (long long)blockIdx.x * blockDim.x + threadIdx.x;
    if (i >= n4) return;
    fv4 f = ((const fv4*)src)[i];
    bv4 b = __builtin_convertvector(f, bv4);
    ((uint2*)dst)[i] = __builtin_bit_cast(uint2, b);
}

// h[n,:] = x_emb1[x[n,0]] + x_emb2[x[n,1]]   (float4 per thread)
__global__ void embed_kernel(const int* __restrict__ x, const float* __restrict__ e1,
                             const float* __restrict__ e2, float* __restrict__ h, int N) {
    long long idx = (long long)blockIdx.x * blockDim.x + threadIdx.x;
    if (idx >= (long long)N * 64) return;
    int n  = (int)(idx >> 6);
    int c4 = (int)(idx & 63) << 2;
    int i0 = x[2 * n], i1 = x[2 * n + 1];
    float4 a = *(const float4*)(e1 + (size_t)i0 * 256 + c4);
    float4 b = *(const float4*)(e2 + (size_t)i1 * 256 + c4);
    *(float4*)(h + (size_t)n * 256 + c4) =
        make_float4(a.x + b.x, a.y + b.y, a.z + b.z, a.w + b.w);
}

// dst[l][n][k] = bf16(src[l][k][n])  -- transpose + convert weights, done once
__global__ void wconvT(const float* __restrict__ src, bfraw* __restrict__ dst,
                       int L, int K, int Nn) {
    int idx = blockIdx.x * blockDim.x + threadIdx.x;
    int total = L * K * Nn;
    if (idx >= total) return;
    int l   = idx / (K * Nn);
    int rem = idx - l * (K * Nn);
    int n   = rem / K;
    int k   = rem - n * K;
    dst[idx] = f2bf(src[(size_t)l * K * Nn + (size_t)k * Nn + n]);
}

// one wave per edge row; lane handles 8 channels (float4 x2 + 8 L2 atomics)
__global__ void scatter_kernel(const float* __restrict__ h, const int* __restrict__ ei,
                               const int* __restrict__ ea, const float* __restrict__ t1,
                               const float* __restrict__ t2, float* __restrict__ agg,
                               int E, int N) {
    int eid = blockIdx.x * 8 + (threadIdx.x >> 5);
    if (eid >= E + N) return;
    int lane = threadIdx.x & 31;
    int s, d, a0, a1;
    if (eid < E) { s = ei[eid]; d = ei[E + eid]; a0 = ea[2 * eid]; a1 = ea[2 * eid + 1]; }
    else         { s = eid - E; d = s; a0 = 4; a1 = 0; }  // self-loop attr = (4,0)
    int c = lane << 3;
    const float* hs = h  + (size_t)s * 256 + c;
    const float* p1 = t1 + (size_t)a0 * 256 + c;
    const float* p2 = t2 + (size_t)a1 * 256 + c;
    float*       ag = agg + (size_t)d * 256 + c;
    float4 h0 = *(const float4*)hs,       h1 = *(const float4*)(hs + 4);
    float4 e0 = *(const float4*)p1,       e1 = *(const float4*)(p1 + 4);
    float4 f0 = *(const float4*)p2,       f1 = *(const float4*)(p2 + 4);
    atomicAdd(ag + 0, h0.x + e0.x + f0.x);
    atomicAdd(ag + 1, h0.y + e0.y + f0.y);
    atomicAdd(ag + 2, h0.z + e0.z + f0.z);
    atomicAdd(ag + 3, h0.w + e0.w + f0.w);
    atomicAdd(ag + 4, h1.x + e1.x + f1.x);
    atomicAdd(ag + 5, h1.y + e1.y + f1.y);
    atomicAdd(ag + 6, h1.z + e1.z + f1.z);
    atomicAdd(ag + 7, h1.w + e1.w + f1.w);
}

// ------------------------------ WMMA GEMM -----------------------------------
// C[M,NTOT] = act( A[M,KTOT] * BT[NTOT,KTOT]^T + bias ),  A/BT in bf16.
// Block: 256 thr = 8 waves; block tile 128x64; wave tile 32x32 (2x2 WMMAs).
// K stepped by 32 through a DOUBLE-BUFFERED async global->LDS pipeline:
// issue copies for step s+1, s_wait_asynccnt<=3 retires step s's copies
// (async loads complete in order), so the async engine overlaps the WMMAs.
// M must be a multiple of 128 (padded) so EXEC is all-ones for WMMA.
template <int KTOT, int NTOT, bool RELU_BF16_OUT>
__global__ void gemm_kernel(const bfraw* __restrict__ A, const bfraw* __restrict__ BT,
                            const float* __restrict__ bias, void* __restrict__ C) {
    __shared__ __align__(16) bfraw ldsA[2][128 * 32];   // 2 x 8 KB
    __shared__ __align__(16) bfraw ldsB[2][64 * 32];    // 2 x 4 KB

    const int tid  = threadIdx.x;
    const int wave = tid >> 5, lane = tid & 31;
    const int wm = wave >> 1, wn = wave & 1;          // 4 x 2 wave grid
    const int lane15 = lane & 15, khalf = lane >> 4;
    const int mBase = blockIdx.x * 128;
    const int nBase = blockIdx.y * 64;

    // per-thread staging coordinates
    const int am = tid >> 1,  akq = (tid & 1) * 16;   // A: 32B per thread
    const int bn = tid >> 2,  bkq = (tid & 3) * 8;    // B: 16B per thread

    auto stage = [&](int k0, int buf) {
        const bfraw* gA = A  + (size_t)(mBase + am) * KTOT + k0 + akq;
        const bfraw* gB = BT + (size_t)(nBase + bn) * KTOT + k0 + bkq;
        bfraw* lA = &ldsA[buf][am * 32 + akq];
        bfraw* lB = &ldsB[buf][bn * 32 + bkq];
#if HAVE_ASYNC_LDS
        async_cp16(gA,     lA);
        async_cp16(gA + 8, lA + 8);
        async_cp16(gB,     lB);
#else
        uint4 qa0 = ((const uint4*)gA)[0];
        uint4 qa1 = ((const uint4*)gA)[1];
        uint4 qb  = *(const uint4*)gB;
        ((uint4*)lA)[0] = qa0; ((uint4*)lA)[1] = qa1;
        *(uint4*)lB = qb;
#endif
    };

    v8f acc[2][2];
    const v8f vzero = {0.f, 0.f, 0.f, 0.f, 0.f, 0.f, 0.f, 0.f};
    acc[0][0] = vzero; acc[0][1] = vzero; acc[1][0] = vzero; acc[1][1] = vzero;

    constexpr int STEPS = KTOT / 32;
    stage(0, 0);                                       // prologue

    for (int s = 0; s < STEPS; ++s) {
        const int buf = s & 1;
        if (s + 1 < STEPS) {
            stage((s + 1) * 32, buf ^ 1);              // prefetch next tile
#if HAVE_ASYNC_LDS
            __builtin_amdgcn_s_wait_asynccnt(3);       // retire current tile only
#endif
        } else {
#if HAVE_ASYNC_LDS
            __builtin_amdgcn_s_wait_asynccnt(0);
#endif
        }
        __syncthreads();

        // ---- fragments per ISA 16-bit A/B layouts ----
        FragAB aF[2], bF[2];
        #pragma unroll
        for (int i = 0; i < 2; ++i) {
            int r = wm * 32 + i * 16 + lane15;
            const bfraw* p = &ldsA[buf][r * 32 + khalf * 8];
            aF[i].q[0] = *(const uint4*)p;          // k = khalf*8 + 0..7
            aF[i].q[1] = *(const uint4*)(p + 16);   // k = 16 + khalf*8 + 0..7
        }
        #pragma unroll
        for (int j = 0; j < 2; ++j) {
            int c = wn * 32 + j * 16 + lane15;
            const bfraw* p = &ldsB[buf][c * 32 + khalf * 8];
            bF[j].q[0] = *(const uint4*)p;          // k = khalf*8 + 0..7
            bF[j].q[1] = *(const uint4*)(p + 16);   // k = 16 + khalf*8 + 0..7
        }
        #pragma unroll
        for (int i = 0; i < 2; ++i)
            #pragma unroll
            for (int j = 0; j < 2; ++j)
                acc[i][j] = __builtin_amdgcn_wmma_f32_16x16x32_bf16(
                    false, aF[i].v, false, bF[j].v, (short)0, acc[i][j], false, false);

        __syncthreads();   // protect buf from step s+2's staging
    }

    // ---- epilogue: bias (+relu -> bf16) or plain f32 store ----
    #pragma unroll
    for (int i = 0; i < 2; ++i) {
        #pragma unroll
        for (int j = 0; j < 2; ++j) {
            int col = nBase + wn * 32 + j * 16 + lane15;
            float bv = bias[col];
            int rbase = mBase + wm * 32 + i * 16 + khalf * 8;
            #pragma unroll
            for (int v = 0; v < 8; ++v) {
                float val = acc[i][j][v] + bv;
                size_t o = (size_t)(rbase + v) * NTOT + col;
                if (RELU_BF16_OUT) {
                    val = fmaxf(val, 0.f);
                    ((bfraw*)C)[o] = f2bf(val);
                } else {
                    ((float*)C)[o] = val;
                }
            }
        }
    }
}

// -------------------------------- BatchNorm ---------------------------------

__global__ void bn_stats(const float* __restrict__ h2, float* __restrict__ stats, int N) {
    int c = threadIdx.x;                       // 256 columns
    int rpb = (N + gridDim.x - 1) / gridDim.x;
    int r0 = blockIdx.x * rpb;
    int r1 = r0 + rpb; if (r1 > N) r1 = N;
    float s = 0.f, ss = 0.f;
    for (int r = r0; r < r1; ++r) {
        float v = h2[(size_t)r * 256 + c];     // coalesced across threads
        s += v; ss += v * v;
    }
    atomicAdd(&stats[c], s);
    atomicAdd(&stats[256 + c], ss);
}

__global__ void bn_apply(const float* __restrict__ h2, const float* __restrict__ stats,
                         const float* __restrict__ g, const float* __restrict__ b,
                         float* __restrict__ out, int N, int relu) {
    long long idx = (long long)blockIdx.x * blockDim.x + threadIdx.x;  // N*64 fv4's
    if (idx >= (long long)N * 64) return;
    int c4 = (int)(idx & 63) << 2;
    float inv = 1.f / (float)N;
    fv4 sum = *(const fv4*)(stats + c4);
    fv4 sq  = *(const fv4*)(stats + 256 + c4);
    fv4 gv  = *(const fv4*)(g + c4);
    fv4 bv  = *(const fv4*)(b + c4);
    fv4 mean = sum * inv;
    fv4 var  = sq * inv - mean * mean;
    fv4 scale;
    scale.x = gv.x * rsqrtf(var.x + 1e-5f);
    scale.y = gv.y * rsqrtf(var.y + 1e-5f);
    scale.z = gv.z * rsqrtf(var.z + 1e-5f);
    scale.w = gv.w * rsqrtf(var.w + 1e-5f);
    fv4 x = ((const fv4*)h2)[idx];
    fv4 r = (x - mean) * scale + bv;
    if (relu) {
        r.x = fmaxf(r.x, 0.f); r.y = fmaxf(r.y, 0.f);
        r.z = fmaxf(r.z, 0.f); r.w = fmaxf(r.w, 0.f);
    }
    ((fv4*)out)[idx] = r;
}

// --------------------------------- driver -----------------------------------

extern "C" void kernel_launch(void* const* d_in, const int* in_sizes, int n_in,
                              void* d_out, int out_size, void* d_ws, size_t ws_size,
                              hipStream_t stream) {
    const int*   x        = (const int*)  d_in[0];
    const int*   ei       = (const int*)  d_in[1];
    const int*   ea       = (const int*)  d_in[2];
    const float* x_emb1   = (const float*)d_in[3];
    const float* x_emb2   = (const float*)d_in[4];
    const float* edge_e1  = (const float*)d_in[5];   // [5][5][256]
    const float* edge_e2  = (const float*)d_in[6];   // [5][3][256]
    const float* W1       = (const float*)d_in[7];   // [5][256][512]
    const float* b1       = (const float*)d_in[8];   // [5][512]
    const float* W2       = (const float*)d_in[9];   // [5][512][256]
    const float* b2       = (const float*)d_in[10];  // [5][256]
    const float* bn_g     = (const float*)d_in[11];  // [5][256]
    const float* bn_b     = (const float*)d_in[12];  // [5][256]
    float* out = (float*)d_out;

    const int N = in_sizes[0] / 2;                 // 100000
    const int E = in_sizes[1] / 2;                 // 300000
    const int M_PAD = ((N + 127) / 128) * 128;     // EXEC all-ones for WMMA tiles

    // workspace carve-out (~360 MB)
    char* ws = (char*)d_ws;
    size_t off = 0;
    auto carve = [&](size_t bytes) -> void* {
        void* p = ws + off; off += (bytes + 255) & ~(size_t)255; return p;
    };
    float* h     = (float*)carve((size_t)M_PAD * 256 * 4);
    float* agg   = (float*)carve((size_t)M_PAD * 256 * 4);   // aliased as h2
    bfraw* aggbf = (bfraw*)carve((size_t)M_PAD * 256 * 2);   // bf16 copy of agg
    bfraw* hid   = (bfraw*)carve((size_t)M_PAD * 512 * 2);
    bfraw* W1T   = (bfraw*)carve((size_t)5 * 512 * 256 * 2); // [l][512][256]
    bfraw* W2T   = (bfraw*)carve((size_t)5 * 256 * 512 * 2); // [l][256][512]
    float* stats = (float*)carve(512 * 4);
    float* h2 = agg;   // agg dead after aggbf conversion -> reuse for gemm2 output

    // one-time weight transpose+convert to bf16
    wconvT<<<(5 * 512 * 256 + 255) / 256, 256, 0, stream>>>(W1, W1T, 5, 256, 512);
    wconvT<<<(5 * 256 * 512 + 255) / 256, 256, 0, stream>>>(W2, W2T, 5, 512, 256);

    // input embedding
    embed_kernel<<<(int)(((long long)N * 64 + 255) / 256), 256, 0, stream>>>(
        x, x_emb1, x_emb2, h, N);

    const long long aggF = (long long)M_PAD * 256;
    for (int l = 0; l < 5; ++l) {
        zero_f4<<<(int)((aggF / 4 + 255) / 256), 256, 0, stream>>>((float4*)agg, aggF / 4);
        zero_f4<<<1, 128, 0, stream>>>((float4*)stats, 128);

        scatter_kernel<<<(E + N + 7) / 8, 256, 0, stream>>>(
            h, ei, ea, edge_e1 + (size_t)l * 5 * 256, edge_e2 + (size_t)l * 3 * 256,
            agg, E, N);

        // agg (f32) -> aggbf (bf16) for the WMMA A operand
        f32_to_bf16_vec<<<(int)((aggF / 4 + 255) / 256), 256, 0, stream>>>(
            agg, aggbf, aggF / 4);

        // hid = relu(aggbf @ W1 + b1)  -> bf16
        gemm_kernel<256, 512, true><<<dim3(M_PAD / 128, 8), 256, 0, stream>>>(
            aggbf, W1T + (size_t)l * 512 * 256, b1 + (size_t)l * 512, hid);

        // h2 = hid @ W2 + b2  -> f32
        gemm_kernel<512, 256, false><<<dim3(M_PAD / 128, 4), 256, 0, stream>>>(
            hid, W2T + (size_t)l * 256 * 512, b2 + (size_t)l * 256, h2);

        bn_stats<<<256, 256, 0, stream>>>(h2, stats, N);

        float* dst = (l == 4) ? out : h;
        bn_apply<<<(int)(((long long)N * 64 + 255) / 256), 256, 0, stream>>>(
            h2, stats, bn_g + (size_t)l * 256, bn_b + (size_t)l * 256, dst, N,
            (l < 4) ? 1 : 0);
    }
}
